// EnhancedTTTRNNCell_46909632807527
// MI455X (gfx1250) — compile-verified, matrix operand order checked by reading, not run on hardware
//
#include <hip/hip_runtime.h>
#include <hip/hip_bf16.h>
#include <math.h>

// ---------------- problem constants ----------------
#define U_    256
#define HID   1024
#define BSZ   512
#define LSEQ  128
#define FEAT  128
#define NROWS (BSZ*(LSEQ-1))   /* 65024 */
#define LN_EPSF 1e-3f
#define LRF     0.01f

// ---------------- WMMA types ----------------
typedef __attribute__((ext_vector_type(16))) __bf16 v16bf;
typedef __attribute__((ext_vector_type(8)))  float  v8f;

union FragBF { v16bf v; unsigned int u[8]; uint4 q[2]; };

// ---------------- scalar helpers ----------------
__device__ __forceinline__ unsigned short f2bf_rn(float f) {
  union { float f; unsigned int u; } c; c.f = f;
  unsigned int u = c.u;
  if ((u & 0x7fffffffu) > 0x7f800000u) return (unsigned short)((u >> 16) | 0x40u);
  return (unsigned short)((u + 0x7fffu + ((u >> 16) & 1u)) >> 16);
}
__device__ __forceinline__ unsigned int pkpair(float a, float b) {
  return (unsigned int)f2bf_rn(a) | ((unsigned int)f2bf_rn(b) << 16);
}
__device__ __forceinline__ float bf2f(unsigned short h) {
  union { unsigned int u; float f; } c; c.u = ((unsigned int)h) << 16; return c.f;
}
__device__ __forceinline__ float gelu_exact(float x) {
  return 0.5f * x * (1.0f + erff(x * 0.7071067811865475f));
}
__device__ __forceinline__ float gelu_grad(float x) {
  float cdf = 0.5f * (1.0f + erff(x * 0.7071067811865475f));
  float pdf = 0.3989422804014327f * __expf(-0.5f * x * x);
  return cdf + x * pdf;
}

// Contiguous fragment load: lane's 16 bf16 live at base+[0..7] and base+[16..23].
__device__ __forceinline__ void load_frag_contig(const unsigned short* base, FragBF& f) {
  const uint4* p = (const uint4*)base;   // 16B-aligned by construction
  f.q[0] = p[0];
  f.q[1] = p[2];
}
// Packed fragment load: lane's 16 bf16 fully contiguous (32B).
__device__ __forceinline__ void load_frag_packed(const unsigned int* base, FragBF& f) {
  const uint4* p = (const uint4*)base;
  f.q[0] = p[0];
  f.q[1] = p[1];
}

// ---------------- templated bf16 WMMA GEMM ----------------
// C[M,N] = A*B (+bias).
// PACKA: A pre-packed in per-lane fragment order [mtile][ktile][lane][16 bf16].
// TB:    B[k,n] = Bph[n*ldb+k] (row-major transposed operand, contiguous in k).
// PACKB: B pre-packed fragment order [ntile][ktile][lane][16 bf16].
// ACT: 0 none; 1 outF=pre-act, outB/outPk=gelu(pre); 2 acc*=gelu'(gateZ).
// outPk (optional): fragment-packed *transposed* copy of the output, i.e. the
// output laid out as a packed operand with Kdim=M, Ndim=N (one uint4/lane).
// Block: 256 thr = 8 waves = 2(M) x 4(N) tiles of 16x16 -> 32x64 per block.
// K must be a multiple of 64 (2-step unroll, dual accumulators).
template <int PACKA, int TB, int PACKB, int ACT>
__global__ void wmma_gemm_kernel(const unsigned short* __restrict__ A, int lda,
                                 const unsigned short* __restrict__ B, int ldb,
                                 const float* __restrict__ bias,
                                 const float* __restrict__ gateZ,
                                 float* __restrict__ outF, unsigned short* __restrict__ outB,
                                 unsigned int* __restrict__ outPk,
                                 int M, int N, int K) {
  const int lane = threadIdx.x & 31;
  const int wave = threadIdx.x >> 5;
  const int wM   = wave >> 2;
  const int tileM = blockIdx.y * 32 + wM * 16;
  const int tileN = blockIdx.x * 64 + (wave & 3) * 16;
  if (tileM >= M || tileN >= N) return;
  const int half = lane >> 4;
  const int mn   = lane & 15;

  const unsigned short* aBase = nullptr;
  const unsigned int*   aPk   = nullptr;
  if constexpr (PACKA) aPk = (const unsigned int*)A + (((size_t)(tileM >> 4) * (K >> 5)) * 32 + lane) * 8;
  else                 aBase = A + (size_t)(tileM + mn) * lda + half * 8;
  const unsigned short* bBase = nullptr;
  const unsigned int*   bPk   = nullptr;
  if constexpr (PACKB) bPk = (const unsigned int*)B + (((size_t)(tileN >> 4) * (K >> 5)) * 32 + lane) * 8;
  else if constexpr (TB) bBase = B + (size_t)(tileN + mn) * ldb + half * 8;
  else                   bBase = B + (size_t)(half * 8) * ldb + (tileN + mn);

  v8f acc0 = {}, acc1 = {};
  for (int k0 = 0; k0 < K; k0 += 64) {
    FragBF a0, a1, b0, b1;
    if constexpr (PACKA) {
      load_frag_packed(aPk + (size_t)(k0 >> 5) * 256, a0);
      load_frag_packed(aPk + (size_t)((k0 >> 5) + 1) * 256, a1);
    } else {
      load_frag_contig(aBase + k0, a0);
      load_frag_contig(aBase + k0 + 32, a1);
    }
    if constexpr (PACKB) {
      load_frag_packed(bPk + (size_t)(k0 >> 5) * 256, b0);
      load_frag_packed(bPk + (size_t)((k0 >> 5) + 1) * 256, b1);
    } else {
      load_frag_contig(bBase + k0, b0);
      load_frag_contig(bBase + k0 + 32, b1);
    }
    acc0 = __builtin_amdgcn_wmma_f32_16x16x32_bf16(false, a0.v, false, b0.v, (short)0, acc0, false, false);
    acc1 = __builtin_amdgcn_wmma_f32_16x16x32_bf16(false, a1.v, false, b1.v, (short)0, acc1, false, false);
  }

  // Epilogue. C/D layout: VGPR i -> row i + 8*half, col = lane&15.
  float vals[8];
#pragma unroll
  for (int i = 0; i < 8; ++i) {
    const int row = tileM + i + half * 8;
    const int col = tileN + mn;
    float v = acc0[i] + acc1[i];
    if (bias) v += bias[col];
    const size_t idx = (size_t)row * N + col;
    if constexpr (ACT == 1) {
      if (outF) outF[idx] = v;           // pre-activation (Z)
      v = gelu_exact(v);
    } else if constexpr (ACT == 2) {
      v *= gelu_grad(gateZ[idx]);
      if (outF) outF[idx] = v;
    } else {
      if (outF) outF[idx] = v;
    }
    if (outB) outB[idx] = f2bf_rn(v);
    vals[i] = v;
  }
  if (outPk) {
    // fragment-packed transposed copy: k-tile index = tileM>>5 (= blockIdx.y),
    // n-tile = tileN>>4, fragment lane = half*16+mn = hardware lane,
    // waveM selects u32 slots [wM*4 .. wM*4+3]  -> single 16B store per lane.
    uint4 q;
    q.x = pkpair(vals[0], vals[1]);
    q.y = pkpair(vals[2], vals[3]);
    q.z = pkpair(vals[4], vals[5]);
    q.w = pkpair(vals[6], vals[7]);
    unsigned int* dst = outPk +
        ((((size_t)(tileN >> 4) * (M >> 5)) + (tileM >> 5)) * 32 + lane) * 8 + wM * 4;
    *(uint4*)dst = q;
  }
}

// ---------------- weight pack: f32 [K,N] -> per-lane fragment-B layout ----------------
__global__ void pack_b_f32_kernel(const float* __restrict__ B, unsigned int* __restrict__ P,
                                  int K, int N) {
  const size_t total = (size_t)K * N / 2;  // one u32 (bf16 pair) per thread iter
  const int ktiles = K >> 5;
  for (size_t i = (size_t)blockIdx.x * blockDim.x + threadIdx.x; i < total;
       i += (size_t)gridDim.x * blockDim.x) {
    const int u    = (int)(i & 7);
    const int lane = (int)((i >> 3) & 31);
    const size_t tile = i >> 8;
    const int kt = (int)(tile % ktiles);
    const int nt = (int)(tile / ktiles);
    const int kk = ((u < 4) ? (2 * u) : (16 + 2 * (u - 4))) + (lane >> 4) * 8;
    const int k = kt * 32 + kk;
    const int n = nt * 16 + (lane & 15);
    P[i] = pkpair(B[(size_t)k * N + n], B[(size_t)(k + 1) * N + n]);
  }
}

// ---------------- elementwise / reduction kernels ----------------
__global__ void f32_to_bf16_kernel(const float* __restrict__ in, unsigned short* __restrict__ out, size_t n) {
  for (size_t i = (size_t)blockIdx.x * blockDim.x + threadIdx.x; i < n;
       i += (size_t)gridDim.x * blockDim.x)
    out[i] = f2bf_rn(in[i]);
}

__global__ void build_buf_kernel(const float* __restrict__ seq, const float* __restrict__ proj,
                                 float* __restrict__ bufo) {
  const size_t total = (size_t)BSZ * LSEQ * U_;
  for (size_t i = (size_t)blockIdx.x * blockDim.x + threadIdx.x; i < total;
       i += (size_t)gridDim.x * blockDim.x) {
    size_t b = i / ((size_t)LSEQ * U_);
    size_t rem = i % ((size_t)LSEQ * U_);
    size_t t = rem / U_, c = rem % U_;
    bufo[i] = (t < LSEQ - 1) ? seq[b * (size_t)LSEQ * U_ + (t + 1) * U_ + c]
                             : proj[b * U_ + c];
  }
}

__global__ void gather_x_kernel(const float* __restrict__ bufo, unsigned short* __restrict__ xbf) {
  const size_t total = (size_t)NROWS * U_;
  for (size_t i = (size_t)blockIdx.x * blockDim.x + threadIdx.x; i < total;
       i += (size_t)gridDim.x * blockDim.x) {
    size_t r = i / U_, c = i % U_;
    size_t b = r / (LSEQ - 1), t = r % (LSEQ - 1);
    xbf[i] = f2bf_rn(bufo[b * (size_t)LSEQ * U_ + t * U_ + c]);
  }
}

__global__ void last_row_kernel(const float* __restrict__ bufo, unsigned short* __restrict__ lastbf) {
  int i = blockIdx.x * blockDim.x + threadIdx.x;
  if (i >= BSZ * U_) return;
  int b = i / U_, c = i % U_;
  lastbf[i] = f2bf_rn(bufo[(size_t)b * LSEQ * U_ + (size_t)(LSEQ - 1) * U_ + c]);
}

__global__ void ln_fwd_kernel(const float* __restrict__ pre, const float* __restrict__ g,
                              const float* __restrict__ beta, float* __restrict__ rec,
                              float* __restrict__ meanv, float* __restrict__ rstdv) {
  const int row = blockIdx.x, c = threadIdx.x;
  const float x = pre[(size_t)row * U_ + c];
  __shared__ float s[U_];
  s[c] = x; __syncthreads();
  for (int off = 128; off > 0; off >>= 1) { if (c < off) s[c] += s[c + off]; __syncthreads(); }
  const float m = s[0] * (1.0f / U_); __syncthreads();
  s[c] = (x - m) * (x - m); __syncthreads();
  for (int off = 128; off > 0; off >>= 1) { if (c < off) s[c] += s[c + off]; __syncthreads(); }
  const float rs = rsqrtf(s[0] * (1.0f / U_) + LN_EPSF);
  rec[(size_t)row * U_ + c] = g[c] * (x - m) * rs + beta[c];
  if (c == 0) { meanv[row] = m; rstdv[row] = rs; }
}

__global__ void ln_bwd_kernel(const float* __restrict__ pre, const float* __restrict__ rec,
                              const float* __restrict__ bufo, const float* __restrict__ g,
                              const float* __restrict__ meanv, const float* __restrict__ rstdv,
                              unsigned short* __restrict__ dpreB) {
  const int row = blockIdx.x, c = threadIdx.x;
  const int b = row / (LSEQ - 1), t = row % (LSEQ - 1);
  const float y = bufo[(size_t)b * LSEQ * U_ + (size_t)(t + 1) * U_ + c];
  const size_t idx = (size_t)row * U_ + c;
  const float dr = (rec[idx] - y) * (2.0f / ((float)NROWS * (float)U_));
  const float a  = g[c] * dr;
  const float xh = (pre[idx] - meanv[row]) * rstdv[row];
  __shared__ float s1[U_], s2[U_];
  s1[c] = a; s2[c] = a * xh; __syncthreads();
  for (int off = 128; off > 0; off >>= 1) {
    if (c < off) { s1[c] += s1[c + off]; s2[c] += s2[c + off]; }
    __syncthreads();
  }
  const float dp = rstdv[row] * (a - s1[0] * (1.0f / U_) - xh * s2[0] * (1.0f / U_));
  dpreB[idx] = f2bf_rn(dp);
}

__global__ void colsum_bf_kernel(const unsigned short* __restrict__ in, float* __restrict__ out,
                                 int rows, int cols) {
  const int col = blockIdx.x * blockDim.x + threadIdx.x;
  if (col >= cols) return;
  int r0 = blockIdx.y * 512;
  int r1 = r0 + 512; if (r1 > rows) r1 = rows;
  float s = 0.0f;
  for (int r = r0; r < r1; ++r) s += bf2f(in[(size_t)r * cols + col]);
  atomicAdd(&out[col], s);
}

__global__ void sgd_kernel(const float* __restrict__ w, const float* __restrict__ gr, float lr,
                           float* __restrict__ outF, int n) {
  int i = blockIdx.x * blockDim.x + threadIdx.x;
  if (i >= n) return;
  outF[i] = w[i] - lr * gr[i];
}

__global__ void contrib_kernel(const float* __restrict__ rec, const float* __restrict__ encp,
                               unsigned short* __restrict__ tmpB) {
  int i = blockIdx.x * blockDim.x + threadIdx.x;
  if (i >= BSZ * U_) return;
  int b = i / U_, c = i % U_;
  float s = 0.0f;
  for (int t = 0; t < LSEQ - 1; ++t) s += rec[((size_t)(b * (LSEQ - 1) + t)) * U_ + c];
  tmpB[i] = f2bf_rn(encp[i] + s * (1.0f / (LSEQ - 1)));
}

// ---------------- host launcher ----------------
extern "C" void kernel_launch(void* const* d_in, const int* in_sizes, int n_in,
                              void* d_out, int out_size, void* d_ws, size_t ws_size,
                              hipStream_t stream) {
  (void)in_sizes; (void)n_in; (void)out_size; (void)ws_size;
  const float* inputs = (const float*)d_in[0];
  /* d_in[1] prev_hidden: unused by reference */
  const float* seq    = (const float*)d_in[2];
  const float* Wp   = (const float*)d_in[3];  const float* bp   = (const float*)d_in[4];
  const float* Wphi = (const float*)d_in[5];  const float* bphi = (const float*)d_in[6];
  const float* Wpsi = (const float*)d_in[7];  const float* bpsi = (const float*)d_in[8];
  const float* W1   = (const float*)d_in[9];  const float* b1   = (const float*)d_in[10];
  const float* W2   = (const float*)d_in[11]; const float* b2   = (const float*)d_in[12];
  const float* Wg   = (const float*)d_in[13]; const float* gb   = (const float*)d_in[14];
  const float* lng  = (const float*)d_in[15]; const float* lnb  = (const float*)d_in[16];
  const float* Wh   = (const float*)d_in[17]; const float* hb   = (const float*)d_in[18];

  float* out  = (float*)d_out;          // [512,256]
  float* bufo = out + (size_t)BSZ * U_; // [512,128,256]

  char* wsp = (char*)d_ws; size_t off = 0;
  auto alloc = [&](size_t bytes) -> void* {
    void* p = wsp + off; off = (off + bytes + 255) & ~(size_t)255; return p;
  };
  // bf16 flat buffers
  unsigned short* inbf   = (unsigned short*)alloc((size_t)BSZ * FEAT * 2);
  unsigned short* Wgbf   = (unsigned short*)alloc((size_t)U_ * U_ * 2);    // d_enc (TB)
  unsigned short* W2bf   = (unsigned short*)alloc((size_t)HID * U_ * 2);   // dZ1 (TB)
  unsigned short* xbf    = (unsigned short*)alloc((size_t)NROWS * U_ * 2);
  unsigned short* phibf  = (unsigned short*)alloc((size_t)NROWS * U_ * 2);
  unsigned short* encbf  = (unsigned short*)alloc((size_t)NROWS * U_ * 2);
  unsigned short* dprebf = (unsigned short*)alloc((size_t)NROWS * U_ * 2);
  unsigned short* dencbf = (unsigned short*)alloc((size_t)NROWS * U_ * 2);
  unsigned short* Hbf    = (unsigned short*)alloc((size_t)NROWS * HID * 2);
  unsigned short* dZ1bf  = (unsigned short*)alloc((size_t)NROWS * HID * 2);
  unsigned short* lastbf = (unsigned short*)alloc((size_t)BSZ * U_ * 2);
  unsigned short* psibf  = (unsigned short*)alloc((size_t)BSZ * U_ * 2);
  unsigned short* Hpbf   = (unsigned short*)alloc((size_t)BSZ * HID * 2);
  unsigned short* tmpbf  = (unsigned short*)alloc((size_t)BSZ * U_ * 2);
  // fragment-packed weights (bf16 pairs as u32)
  unsigned int* WpPk   = (unsigned int*)alloc((size_t)FEAT * U_ * 2);
  unsigned int* WphiPk = (unsigned int*)alloc((size_t)U_ * U_ * 2);
  unsigned int* WpsiPk = (unsigned int*)alloc((size_t)U_ * U_ * 2);
  unsigned int* WgPk   = (unsigned int*)alloc((size_t)U_ * U_ * 2);
  unsigned int* WhPk   = (unsigned int*)alloc((size_t)U_ * U_ * 2);
  unsigned int* W1Pk   = (unsigned int*)alloc((size_t)U_ * HID * 2);
  unsigned int* W2Pk   = (unsigned int*)alloc((size_t)HID * U_ * 2);
  unsigned int* W1nPk  = (unsigned int*)alloc((size_t)U_ * HID * 2);
  unsigned int* W2nPk  = (unsigned int*)alloc((size_t)HID * U_ * 2);
  // fragment-packed transposed activations (emitted by producer epilogues)
  unsigned int* phiPk  = (unsigned int*)alloc((size_t)NROWS * U_ * 2);   // K=NROWS, N=U
  unsigned int* HPk    = (unsigned int*)alloc((size_t)NROWS * HID * 2);  // K=NROWS, N=HID
  unsigned int* dencPk = (unsigned int*)alloc((size_t)NROWS * U_ * 2);
  unsigned int* dZ1Pk  = (unsigned int*)alloc((size_t)NROWS * HID * 2);
  // f32 buffers
  float* projf = (float*)alloc((size_t)BSZ * U_ * 4);
  float* Z1f   = (float*)alloc((size_t)NROWS * HID * 4);
  float* pref  = (float*)alloc((size_t)NROWS * U_ * 4);
  float* recf  = (float*)alloc((size_t)NROWS * U_ * 4);
  float* meanv = (float*)alloc((size_t)NROWS * 4);
  float* rstdv = (float*)alloc((size_t)NROWS * 4);
  float* gW1f  = (float*)alloc((size_t)U_ * HID * 4);
  float* gW2f  = (float*)alloc((size_t)HID * U_ * 4);
  float* gb1f  = (float*)alloc((size_t)HID * 4);
  float* gb2f  = (float*)alloc((size_t)U_ * 4);
  float* W1nf  = (float*)alloc((size_t)U_ * HID * 4);
  float* W2nf  = (float*)alloc((size_t)HID * U_ * 4);
  float* b1nf  = (float*)alloc((size_t)HID * 4);
  float* b2nf  = (float*)alloc((size_t)U_ * 4);
  float* encpf = (float*)alloc((size_t)BSZ * U_ * 4);

  const dim3 blk(256);
  auto cvt = [&](const float* src, unsigned short* dst, size_t n) {
    unsigned int g = (unsigned int)((n + 255) / 256); if (g > 4096u) g = 4096u;
    f32_to_bf16_kernel<<<dim3(g), blk, 0, stream>>>(src, dst, n);
  };
  auto pack = [&](const float* src, unsigned int* dst, int K, int N) {
    size_t n = (size_t)K * N / 2;
    unsigned int g = (unsigned int)((n + 255) / 256); if (g > 4096u) g = 4096u;
    pack_b_f32_kernel<<<dim3(g), blk, 0, stream>>>(src, dst, K, N);
  };

  // 1) weight prep
  pack(Wp, WpPk, FEAT, U_);
  pack(Wphi, WphiPk, U_, U_);
  pack(Wpsi, WpsiPk, U_, U_);
  pack(Wg, WgPk, U_, U_);
  pack(Wh, WhPk, U_, U_);
  pack(W1, W1Pk, U_, HID);
  pack(W2, W2Pk, HID, U_);
  cvt(Wg, Wgbf, (size_t)U_ * U_);
  cvt(W2, W2bf, (size_t)HID * U_);
  cvt(inputs, inbf, (size_t)BSZ * FEAT);

  // 2) proj = inputs @ Wp + bp            [512,256]
  wmma_gemm_kernel<0,0,1,0><<<dim3(U_/64, BSZ/32), blk, 0, stream>>>(
      inbf, FEAT, (const unsigned short*)WpPk, 0, bp, nullptr, projf, nullptr, nullptr, BSZ, U_, FEAT);
  // 3) buf (second output)
  build_buf_kernel<<<dim3(4096), blk, 0, stream>>>(seq, projf, bufo);
  // 4) gather x rows and last row
  gather_x_kernel<<<dim3(4096), blk, 0, stream>>>(bufo, xbf);
  last_row_kernel<<<dim3((BSZ * U_ + 255) / 256), blk, 0, stream>>>(bufo, lastbf);
  // 5) phi = x @ Wphi + bphi              [65024,256]   (+ packed phi^T)
  wmma_gemm_kernel<0,0,1,0><<<dim3(U_/64, NROWS/32), blk, 0, stream>>>(
      xbf, U_, (const unsigned short*)WphiPk, 0, bphi, nullptr, nullptr, phibf, phiPk, NROWS, U_, U_);
  // 6) Z1 = phi @ W1 + b1 ; H = gelu(Z1)  [65024,1024]  (+ packed H^T)
  wmma_gemm_kernel<0,0,1,1><<<dim3(HID/64, NROWS/32), blk, 0, stream>>>(
      phibf, U_, (const unsigned short*)W1Pk, 0, b1, nullptr, Z1f, Hbf, HPk, NROWS, HID, U_);
  // 7) enc = H @ W2 + b2                  [65024,256]
  wmma_gemm_kernel<0,0,1,0><<<dim3(U_/64, NROWS/32), blk, 0, stream>>>(
      Hbf, HID, (const unsigned short*)W2Pk, 0, b2, nullptr, nullptr, encbf, nullptr, NROWS, U_, HID);
  // 8) pre = enc @ Wg + g_bias            [65024,256]
  wmma_gemm_kernel<0,0,1,0><<<dim3(U_/64, NROWS/32), blk, 0, stream>>>(
      encbf, U_, (const unsigned short*)WgPk, 0, gb, nullptr, pref, nullptr, nullptr, NROWS, U_, U_);
  // 9) rec = LN(pre)
  ln_fwd_kernel<<<dim3(NROWS), blk, 0, stream>>>(pref, lng, lnb, recf, meanv, rstdv);
  // 10) d_pre via loss + LN backward
  ln_bwd_kernel<<<dim3(NROWS), blk, 0, stream>>>(pref, recf, bufo, lng, meanv, rstdv, dprebf);
  // 11) d_enc = d_pre @ Wg^T   (TB)       (+ packed d_enc^T)
  wmma_gemm_kernel<0,1,0,0><<<dim3(U_/64, NROWS/32), blk, 0, stream>>>(
      dprebf, U_, Wgbf, U_, nullptr, nullptr, nullptr, dencbf, dencPk, NROWS, U_, U_);
  // 12) grads
  hipMemsetAsync(gb1f, 0, HID * 4, stream);
  hipMemsetAsync(gb2f, 0, U_ * 4, stream);
  colsum_bf_kernel<<<dim3(1, (NROWS + 511) / 512), blk, 0, stream>>>(dencbf, gb2f, NROWS, U_);
  // gW2 = H^T @ d_enc   [1024,256], K=65024  (fully packed operands)
  wmma_gemm_kernel<1,0,1,0><<<dim3(U_/64, HID/32), blk, 0, stream>>>(
      (const unsigned short*)HPk, 0, (const unsigned short*)dencPk, 0,
      nullptr, nullptr, gW2f, nullptr, nullptr, HID, U_, NROWS);
  // d_Z1 = (d_enc @ W2^T) * gelu'(Z1)   [65024,1024]  (TB, ACT2; + packed dZ1^T)
  wmma_gemm_kernel<0,1,0,2><<<dim3(HID/64, NROWS/32), blk, 0, stream>>>(
      dencbf, U_, W2bf, U_, nullptr, Z1f, nullptr, dZ1bf, dZ1Pk, NROWS, HID, U_);
  colsum_bf_kernel<<<dim3(4, (NROWS + 511) / 512), blk, 0, stream>>>(dZ1bf, gb1f, NROWS, HID);
  // gW1 = phi^T @ d_Z1  [256,1024], K=65024  (fully packed operands)
  wmma_gemm_kernel<1,0,1,0><<<dim3(HID/64, U_/32), blk, 0, stream>>>(
      (const unsigned short*)phiPk, 0, (const unsigned short*)dZ1Pk, 0,
      nullptr, nullptr, gW1f, nullptr, nullptr, U_, HID, NROWS);
  // 13) SGD update (lr = 0.01) + re-pack updated weights
  sgd_kernel<<<dim3((U_ * HID + 255) / 256), blk, 0, stream>>>(W1, gW1f, LRF, W1nf, U_ * HID);
  sgd_kernel<<<dim3((HID * U_ + 255) / 256), blk, 0, stream>>>(W2, gW2f, LRF, W2nf, HID * U_);
  sgd_kernel<<<dim3((HID + 255) / 256), blk, 0, stream>>>(b1, gb1f, LRF, b1nf, HID);
  sgd_kernel<<<dim3((U_ + 255) / 256), blk, 0, stream>>>(b2, gb2f, LRF, b2nf, U_);
  pack(W1nf, W1nPk, U_, HID);
  pack(W2nf, W2nPk, HID, U_);
  // 14) psi = buf[:,-1,:] @ Wpsi + bpsi   [512,256]
  wmma_gemm_kernel<0,0,1,0><<<dim3(U_/64, BSZ/32), blk, 0, stream>>>(
      lastbf, U_, (const unsigned short*)WpsiPk, 0, bpsi, nullptr, nullptr, psibf, nullptr, BSZ, U_, U_);
  // 15) encoder(updated params, psi)
  wmma_gemm_kernel<0,0,1,1><<<dim3(HID/64, BSZ/32), blk, 0, stream>>>(
      psibf, U_, (const unsigned short*)W1nPk, 0, b1nf, nullptr, nullptr, Hpbf, nullptr, BSZ, HID, U_);
  wmma_gemm_kernel<0,0,1,0><<<dim3(U_/64, BSZ/32), blk, 0, stream>>>(
      Hpbf, HID, (const unsigned short*)W2nPk, 0, b2nf, nullptr, encpf, nullptr, nullptr, BSZ, U_, HID);
  // 16) tmp = enc_psi + mean_t(rec)
  contrib_kernel<<<dim3((BSZ * U_ + 255) / 256), blk, 0, stream>>>(recf, encpf, tmpbf);
  // 17) out = tmp @ Wh + h_bias
  wmma_gemm_kernel<0,0,1,0><<<dim3(U_/64, BSZ/32), blk, 0, stream>>>(
      tmpbf, U_, (const unsigned short*)WhPk, 0, hb, nullptr, out, nullptr, nullptr, BSZ, U_, U_);
}
